// MultiHeadAttention_14834817040485
// MI455X (gfx1250) — matrix operand order repaired and checked by reading, not compiled, not run on hardware
//
#include <hip/hip_runtime.h>
#include <stdint.h>

// MI455X / gfx1250 multi-head attention, fused-attention pipeline.
//
// Roofline: total ~69 GFLOP; the attn output [32,2048,2048] f32 (537 MB) dominates memory.
// This version touches attn exactly once (the required output write):
//   1) cvt_f32_f16 : Wq,Wk,Wv,Wo -> f16
//   2) qkv_gemm    : Q/K -> [B,H,S,d] f16, V -> [B,H,d,S] f16 (NT layout for attn@V)
//   3) fused_attn  : per wave: 32 q-columns, loop k-tiles of 64:
//        sweep 1: S^T(k,q) = Kh x Qh^T via WMMA; per-LANE online softmax (m,s) since q
//                 is the lane dimension of the C layout; one shfl_xor(16) merge.
//        sweep 2: recompute S^T, p = exp(v-m)/s -> vectorized f32 stores to attn
//                 (k contiguous per lane), register-transpose p into B-fragments via one
//                 half-wave shfl_xor, accumulate ctx^T = V^T x P^T with WMMA.
//        end: scatter ctx^T (f16) into buggy-permuted Y.
//   4) out_gemm    : Y @ Wo^T + bo -> d_out fp32
// All GEMMs: V_WMMA_F32_16X16X32_F16, fp32 accumulate, NT fragment loads, contiguous k order (k = 16*half + s).
// Workspace: 40 MB. Assumption: jax bool mask = 1 byte/element.

#define S_LEN 2048
#define EMB   1024
#define NB    2
#define NH    16
#define HD    64

typedef __attribute__((ext_vector_type(16))) _Float16 v16h;
typedef __attribute__((ext_vector_type(8)))  _Float16 v8h;
typedef __attribute__((ext_vector_type(4)))  _Float16 v4h;
typedef __attribute__((ext_vector_type(8)))  float    v8f;
typedef __attribute__((ext_vector_type(4)))  float    v4f;

// ---------------- f32 -> f16 convert ----------------
__global__ __launch_bounds__(256) void cvt_f32_f16(const float* __restrict__ src,
                                                   _Float16* __restrict__ dst, int n4) {
  int i = blockIdx.x * blockDim.x + threadIdx.x;
  if (i >= n4) return;
  v4f f = ((const v4f*)src)[i];
  v4h h;
#pragma unroll
  for (int j = 0; j < 4; ++j) h[j] = (_Float16)f[j];
  ((v4h*)dst)[i] = h;
}

// ---------------- fragment loaders (contiguous-k layouts, NT storage) ----------------
// A 16x32 f16: lane(r,sel): row=r, K = sel*16 + 0..15 contiguous
__device__ __forceinline__ v16h load_a_f16(const _Float16* __restrict__ p) {
  v8h lo = *(const v8h*)p;
  v8h hi = *(const v8h*)(p + 8);
  v16h a;
#pragma unroll
  for (int j = 0; j < 8; ++j) { a[j] = lo[j]; a[8 + j] = hi[j]; }
  return a;
}
// B 32x16 f16: lane(r,sel): n=r, K = sel*16 + 0..15 contiguous
__device__ __forceinline__ v16h load_b_f16(const _Float16* __restrict__ p) {
  return *(const v16h*)p;
}

// ---------------- wave-level 64x32 NT GEMM cores (for projection GEMMs) ----------------
__device__ __forceinline__ void wmma_tile_f16(const _Float16* __restrict__ A, int lda,
                                              const _Float16* __restrict__ Bm, int ldb,
                                              int kIters, v8f acc[4][2], int lane) {
  const int r = lane & 15, sel = lane >> 4;
  for (int kk = 0; kk < kIters; ++kk) {
    const int k0 = kk << 5;
    v16h a[4];
#pragma unroll
    for (int mt = 0; mt < 4; ++mt)
      a[mt] = load_a_f16(A + (size_t)(mt * 16 + r) * lda + k0 + sel * 16);
    v16h b[2];
#pragma unroll
    for (int nt = 0; nt < 2; ++nt)
      b[nt] = load_b_f16(Bm + (size_t)(nt * 16 + r) * ldb + k0 + sel * 16);
#pragma unroll
    for (int mt = 0; mt < 4; ++mt)
#pragma unroll
      for (int nt = 0; nt < 2; ++nt)
        acc[mt][nt] = __builtin_amdgcn_wmma_f32_16x16x32_f16(
            false, a[mt], false, b[nt], (short)0, acc[mt][nt], false, false);
  }
}

__device__ __forceinline__ void wmma_tile_f32a(const float* __restrict__ A, int lda,
                                               const _Float16* __restrict__ Bm, int ldb,
                                               int kIters, v8f acc[4][2], int lane) {
  const int r = lane & 15, sel = lane >> 4;
  for (int kk = 0; kk < kIters; ++kk) {
    const int k0 = kk << 5;
    v16h a[4];
#pragma unroll
    for (int mt = 0; mt < 4; ++mt) {
      const float* p = A + (size_t)(mt * 16 + r) * lda + k0 + sel * 16;
      v4f f0 = *(const v4f*)p, f1 = *(const v4f*)(p + 4);
      v4f f2 = *(const v4f*)(p + 8), f3 = *(const v4f*)(p + 12);
#pragma unroll
      for (int j = 0; j < 4; ++j) {
        a[mt][j]      = (_Float16)f0[j];
        a[mt][4 + j]  = (_Float16)f1[j];
        a[mt][8 + j]  = (_Float16)f2[j];
        a[mt][12 + j] = (_Float16)f3[j];
      }
    }
    v16h b[2];
#pragma unroll
    for (int nt = 0; nt < 2; ++nt)
      b[nt] = load_b_f16(Bm + (size_t)(nt * 16 + r) * ldb + k0 + sel * 16);
#pragma unroll
    for (int mt = 0; mt < 4; ++mt)
#pragma unroll
      for (int nt = 0; nt < 2; ++nt)
        acc[mt][nt] = __builtin_amdgcn_wmma_f32_16x16x32_f16(
            false, a[mt], false, b[nt], (short)0, acc[mt][nt], false, false);
  }
}

// ---------------- stage 2: fused QKV projections ----------------
__global__ __launch_bounds__(256) void qkv_gemm(
    const float* __restrict__ q, const float* __restrict__ k, const float* __restrict__ v,
    const _Float16* __restrict__ Wq, const _Float16* __restrict__ Wk, const _Float16* __restrict__ Wv,
    const float* __restrict__ bq, const float* __restrict__ bk, const float* __restrict__ bv,
    _Float16* __restrict__ Qh, _Float16* __restrict__ Kh, _Float16* __restrict__ Vt) {
  const int z = blockIdx.z;
  const float*    X    = (z == 0) ? q  : (z == 1) ? k  : v;
  const _Float16* W    = (z == 0) ? Wq : (z == 1) ? Wk : Wv;
  const float*    bias = (z == 0) ? bq : (z == 1) ? bk : bv;
  _Float16*       dst  = (z == 0) ? Qh : (z == 1) ? Kh : Vt;

  const int lane = threadIdx.x & 31, wave = threadIdx.x >> 5;
  const int row0 = blockIdx.x * 256 + (wave >> 1) * 64;
  const int col0 = blockIdx.y * 64 + (wave & 1) * 32;
  const int r = lane & 15, sel = lane >> 4;

  v8f acc[4][2] = {};
  wmma_tile_f32a(X + (size_t)row0 * EMB, EMB, W + (size_t)col0 * EMB, EMB, EMB / 32, acc, lane);

#pragma unroll
  for (int mt = 0; mt < 4; ++mt)
#pragma unroll
    for (int nt = 0; nt < 2; ++nt)
#pragma unroll
      for (int j = 0; j < 8; ++j) {
        int m = row0 + mt * 16 + sel * 8 + j;
        int n = col0 + nt * 16 + r;
        float val = acc[mt][nt][j] + bias[n];
        int b = m >> 11, s = m & (S_LEN - 1);
        int h = n >> 6, dd = n & 63;
        size_t idx = (z == 2)
            ? ((size_t)((b * NH + h) * HD + dd) * S_LEN + s)    // Vt: [B,H,d,S]
            : ((size_t)((b * NH + h) * S_LEN + s) * HD + dd);   // Qh/Kh: [B,H,S,d]
        dst[idx] = (_Float16)val;
      }
}

// ---------------- stage 3: fused scores + softmax + attn@V ----------------
// Wave owns 32 q columns; k-tile = 64 rows of S^T per iteration (32 iterations).
// S^T C-layout: q = nt*16 + (lane&15);  k = mt*16 + sel*8 + j  (sel = lane>>4).

// Compute masked/scaled transposed score tile st[4][2] for k-tile base `kb`.
__device__ __forceinline__ void score_tile(
    const _Float16* __restrict__ Kb, const v16h (&qb)[2][2],
    const uint8_t* __restrict__ mrow0, const uint8_t* __restrict__ mrow1,
    int kb, int r, int sel, v8f st[4][2]) {
#pragma unroll
  for (int mt = 0; mt < 4; ++mt)
#pragma unroll
    for (int nt = 0; nt < 2; ++nt)
      st[mt][nt] = (v8f)(0.0f);
#pragma unroll
  for (int mt = 0; mt < 4; ++mt)
#pragma unroll
    for (int kk = 0; kk < 2; ++kk) {
      v16h a = load_a_f16(Kb + (size_t)(kb + mt * 16 + r) * HD + kk * 32 + sel * 16);
#pragma unroll
      for (int nt = 0; nt < 2; ++nt)
        st[mt][nt] = __builtin_amdgcn_wmma_f32_16x16x32_f16(
            false, a, false, qb[nt][kk], (short)0, st[mt][nt], false, false);
    }
  // scale + mask (mask bytes contiguous in k per lane)
#pragma unroll
  for (int mt = 0; mt < 4; ++mt)
#pragma unroll
    for (int nt = 0; nt < 2; ++nt) {
      const uint8_t* mp = (nt ? mrow1 : mrow0) + kb + mt * 16 + sel * 8;
      uint2 mv = *(const uint2*)mp;
#pragma unroll
      for (int j = 0; j < 8; ++j) {
        uint32_t byte = (j < 4) ? ((mv.x >> (8 * j)) & 0xffu) : ((mv.y >> (8 * (j - 4))) & 0xffu);
        float val = st[mt][nt][j] * 0.125f;               // 1/sqrt(64)
        st[mt][nt][j] = byte ? -1e18f : val;
      }
    }
}

__global__ __launch_bounds__(256) void fused_attn(
    const _Float16* __restrict__ Qh, const _Float16* __restrict__ Kh,
    const _Float16* __restrict__ Vt, const uint8_t* __restrict__ mask,
    float* __restrict__ attn, _Float16* __restrict__ Y) {
  const int bh = blockIdx.z;
  const int lane = threadIdx.x & 31, wave = threadIdx.x >> 5;
  const int r = lane & 15, sel = lane >> 4;
  const int qb0 = (blockIdx.x * 8 + wave) * 32;           // 32 q per wave
  const int b = bh >> 4;

  const _Float16* Qb = Qh + (size_t)bh * S_LEN * HD;
  const _Float16* Kb = Kh + (size_t)bh * S_LEN * HD;
  const _Float16* Vb = Vt + (size_t)bh * HD * S_LEN;
  const int q0 = qb0 + r, q1 = qb0 + 16 + r;              // the two q this lane owns
  const uint8_t* mrow0 = mask + ((size_t)b * S_LEN + q0) * S_LEN;
  const uint8_t* mrow1 = mask + ((size_t)b * S_LEN + q1) * S_LEN;
  float* arow0 = attn + (size_t)bh * S_LEN * S_LEN + (size_t)q0 * S_LEN;
  float* arow1 = attn + (size_t)bh * S_LEN * S_LEN + (size_t)q1 * S_LEN;

  // Persistent Q B-fragments: qb[nt][kk]  (q = qb0+nt*16+r, d-chunk kk)
  v16h qb[2][2];
#pragma unroll
  for (int nt = 0; nt < 2; ++nt)
#pragma unroll
    for (int kk = 0; kk < 2; ++kk)
      qb[nt][kk] = load_b_f16(Qb + (size_t)(qb0 + nt * 16 + r) * HD + kk * 32 + sel * 16);

  // ---- sweep 1: per-lane online softmax stats over k ----
  float m[2] = {-3.0e38f, -3.0e38f};
  float s[2] = {0.f, 0.f};
  for (int kt = 0; kt < S_LEN / 64; ++kt) {
    v8f st[4][2];
    score_tile(Kb, qb, mrow0, mrow1, kt * 64, r, sel, st);
#pragma unroll
    for (int nt = 0; nt < 2; ++nt)
#pragma unroll
      for (int mt = 0; mt < 4; ++mt)
#pragma unroll
        for (int j = 0; j < 8; ++j) {
          float v = st[mt][nt][j];
          float mn = fmaxf(m[nt], v);
          s[nt] = s[nt] * __expf(m[nt] - mn) + __expf(v - mn);
          m[nt] = mn;
        }
  }
  // merge the two half-wave partials (lanes r and r+16 hold the same q, different k)
#pragma unroll
  for (int nt = 0; nt < 2; ++nt) {
    float om = __shfl_xor(m[nt], 16, 32);
    float os = __shfl_xor(s[nt], 16, 32);
    float mn = fmaxf(m[nt], om);
    s[nt] = s[nt] * __expf(m[nt] - mn) + os * __expf(om - mn);
    m[nt] = mn;
  }
  float inv[2] = {1.0f / s[0], 1.0f / s[1]};

  // ---- sweep 2: recompute, write normalized attn, accumulate ctx^T = V^T x P^T ----
  v8f ctx[4][2] = {};                                     // dd = mtd*16+sel*8+j, q = nt2*16+r
  for (int kt = 0; kt < S_LEN / 64; ++kt) {
    const int kb = kt * 64;
    v8f st[4][2];
    score_tile(Kb, qb, mrow0, mrow1, kb, r, sel, st);
    // p = exp(v - m) / s; overwrite st; vectorized stores (k contiguous per lane)
#pragma unroll
    for (int nt = 0; nt < 2; ++nt)
#pragma unroll
      for (int mt = 0; mt < 4; ++mt) {
#pragma unroll
        for (int j = 0; j < 8; ++j)
          st[mt][nt][j] = __expf(st[mt][nt][j] - m[nt]) * inv[nt];
        float* ap = (nt ? arow1 : arow0) + kb + mt * 16 + sel * 8;
        v4f lo, hi;
#pragma unroll
        for (int j = 0; j < 4; ++j) { lo[j] = st[mt][nt][j]; hi[j] = st[mt][nt][4 + j]; }
        *(v4f*)ap = lo;
        *(v4f*)(ap + 4) = hi;
      }
    // ctx^T accumulation: two K=32 chunks per k-tile
#pragma unroll
    for (int kkc = 0; kkc < 2; ++kkc) {
      v16h aV[4];
#pragma unroll
      for (int mtd = 0; mtd < 4; ++mtd)
        aV[mtd] = load_a_f16(Vb + (size_t)(mtd * 16 + r) * S_LEN + kb + kkc * 32 + sel * 16);
#pragma unroll
      for (int nt2 = 0; nt2 < 2; ++nt2) {
        // register transpose of p into the B-fragment layout:
        // B-frag lane(r,sel) needs k = kb + kkc*32 + sel*16 + 0..15  (mtX = kkc*2+sel);
        // own half has 8 of them, partner lane (xor 16) has the other 8.
        v8f own  = sel ? st[kkc * 2 + 1][nt2] : st[kkc * 2][nt2];
        v8f send = sel ? st[kkc * 2][nt2]     : st[kkc * 2 + 1][nt2];
        v8f recv;
#pragma unroll
        for (int j = 0; j < 8; ++j) recv[j] = __shfl_xor(send[j], 16, 32);
        v16h bf;
#pragma unroll
        for (int j = 0; j < 8; ++j) {
          float loj = sel ? recv[j] : own[j];
          float hij = sel ? own[j] : recv[j];
          bf[j]     = (_Float16)loj;
          bf[8 + j] = (_Float16)hij;
        }
#pragma unroll
        for (int mtd = 0; mtd < 4; ++mtd)
          ctx[mtd][nt2] = __builtin_amdgcn_wmma_f32_16x16x32_f16(
              false, aV[mtd], false, bf, (short)0, ctx[mtd][nt2], false, false);
      }
    }
  }

  // ---- scatter ctx^T into buggy-permuted Y:  bh = h2*B + b2 ----
  const int h2 = bh >> 1, b2 = bh & 1;
#pragma unroll
  for (int mtd = 0; mtd < 4; ++mtd)
#pragma unroll
    for (int nt2 = 0; nt2 < 2; ++nt2)
#pragma unroll
      for (int j = 0; j < 8; ++j) {
        int dd = mtd * 16 + sel * 8 + j;
        int sq = qb0 + nt2 * 16 + r;
        Y[((size_t)(h2 * S_LEN + sq) * NB + b2) * HD + dd] = (_Float16)ctx[mtd][nt2][j];
      }
}

// ---------------- stage 4: out = Y @ Wo^T + bo ----------------
__global__ __launch_bounds__(256) void out_gemm(
    const _Float16* __restrict__ Y, const _Float16* __restrict__ Wo,
    const float* __restrict__ bo, float* __restrict__ out) {
  const int lane = threadIdx.x & 31, wave = threadIdx.x >> 5;
  const int row0 = blockIdx.x * 256 + (wave >> 1) * 64;
  const int col0 = blockIdx.y * 64 + (wave & 1) * 32;
  const int r = lane & 15, sel = lane >> 4;

  v8f acc[4][2] = {};
  wmma_tile_f16(Y + (size_t)row0 * EMB, EMB, Wo + (size_t)col0 * EMB, EMB, EMB / 32, acc, lane);

#pragma unroll
  for (int mt = 0; mt < 4; ++mt)
#pragma unroll
    for (int nt = 0; nt < 2; ++nt)
#pragma unroll
      for (int j = 0; j < 8; ++j) {
        int mm = row0 + mt * 16 + sel * 8 + j;
        int nn = col0 + nt * 16 + r;
        out[(size_t)mm * EMB + nn] = acc[mt][nt][j] + bo[nn];
      }
}

// ---------------- launcher ----------------
extern "C" void kernel_launch(void* const* d_in, const int* in_sizes, int n_in,
                              void* d_out, int out_size, void* d_ws, size_t ws_size,
                              hipStream_t stream) {
  (void)in_sizes; (void)n_in; (void)out_size; (void)ws_size;

  const float*   q    = (const float*)d_in[0];
  const float*   k    = (const float*)d_in[1];
  const float*   v    = (const float*)d_in[2];
  const uint8_t* mask = (const uint8_t*)d_in[3];
  // d_in[4] = num_heads == 16 (compile-time constant here)
  const float* Wq = (const float*)d_in[5];
  const float* bq = (const float*)d_in[6];
  const float* Wk = (const float*)d_in[7];
  const float* bk = (const float*)d_in[8];
  const float* Wv = (const float*)d_in[9];
  const float* bv = (const float*)d_in[10];
  const float* Wo = (const float*)d_in[11];
  const float* bo = (const float*)d_in[12];

  float* out  = (float*)d_out;                                   // [B,S,E]   fp32
  float* attn = out + (size_t)NB * S_LEN * EMB;                  // [B*H,S,S] fp32

  const size_t WELEM = (size_t)EMB * EMB;
  const size_t TELEM = (size_t)NB * NH * S_LEN * HD;
  _Float16* wsh  = (_Float16*)d_ws;
  _Float16* Wq16 = wsh;
  _Float16* Wk16 = Wq16 + WELEM;
  _Float16* Wv16 = Wk16 + WELEM;
  _Float16* Wo16 = Wv16 + WELEM;
  _Float16* Qh   = Wo16 + WELEM;
  _Float16* Kh   = Qh + TELEM;
  _Float16* Vt   = Kh + TELEM;
  _Float16* Y    = Vt + TELEM;                                   // 40 MB total

  {
    int n4 = (int)(WELEM / 4);
    dim3 g((n4 + 255) / 256), bdim(256);
    cvt_f32_f16<<<g, bdim, 0, stream>>>(Wq, Wq16, n4);
    cvt_f32_f16<<<g, bdim, 0, stream>>>(Wk, Wk16, n4);
    cvt_f32_f16<<<g, bdim, 0, stream>>>(Wv, Wv16, n4);
    cvt_f32_f16<<<g, bdim, 0, stream>>>(Wo, Wo16, n4);
  }
  // QKV projections (M=4096, N=1024, K=1024)
  qkv_gemm<<<dim3(16, 16, 3), 256, 0, stream>>>(q, k, v, Wq16, Wk16, Wv16,
                                                bq, bk, bv, Qh, Kh, Vt);
  // fused scores+softmax+attn@V: 8 q-block WGs x 32 (b,h)
  fused_attn<<<dim3(8, 1, 32), 256, 0, stream>>>(Qh, Kh, Vt, mask, attn, Y);
  // final projection
  out_gemm<<<dim3(16, 16, 1), 256, 0, stream>>>(Y, Wo16, bo, out);
}
